// Attention_1683627180341
// MI455X (gfx1250) — matrix operand order repaired and checked by reading, not compiled, not run on hardware
//
#include <hip/hip_runtime.h>
#include <hip/hip_bf16.h>

// ---------------------------------------------------------------------------
// MI455X (gfx1250) attention: all matmuls on v_wmma_f32_16x16x32_f16.
// B=2, N=2048, D=1024, H=16, DH=64, SCALE=1/8.
// Pipeline:
//   convert/transpose -> GEMM(qk) -> GEMM(v) -> GEMM(xxt) -> scatter ->
//   fused per-(b,rowtile) attention (320KB-LDS resident softmax + head loop,
//   q tile staged via TENSOR_LOAD_TO_LDS) ->
//   GEMM(out = y @ WoT, f32 overwrite) -> GEMM(out += asum @ x, f32 accum)
// Workspace use ~88 MB (fits in 192 MB L2).
// ---------------------------------------------------------------------------

typedef __attribute__((ext_vector_type(16))) _Float16 v16h;
typedef __attribute__((ext_vector_type(8)))  _Float16 v8h;
typedef __attribute__((ext_vector_type(8)))  float    v8f;
typedef __attribute__((ext_vector_type(4))) unsigned int u32x4;
typedef __attribute__((ext_vector_type(8))) int         i32x8;
typedef __attribute__((ext_vector_type(4))) int         i32x4;

#define BATCH 2
#define NSEQ  2048
#define DMODEL 1024
#define NHEAD 16
#define HDIM  64
#define MT    32          // attention row tile
#define SCALE 0.125f

#if defined(__has_builtin)
#if __has_builtin(__builtin_amdgcn_tensor_load_to_lds) && __has_builtin(__builtin_amdgcn_s_wait_tensorcnt)
#define USE_TDM 1
#endif
#endif
#ifndef USE_TDM
#define USE_TDM 0
#endif

// Build a 16-half WMMA fragment from two 16-byte chunks.
// A-fragment (16x32 f16): slots 0..7 = A[m][base .. base+7], slots 8..15 = A[m][base+16 .. base+23]
//   -> ld_frag16(p, p+16) with p = &A[m][kstep*32 + half*8]
// B-fragment (32x16 f16): lane -> n = n0 + l16, slots s<8 = B[kbase+half*8+s][n], s>=8 = B[kbase+16+half*8+(s-8)][n]
//   -> strided per-element gather (same table k order as A)
static __device__ __forceinline__ v16h ld_frag16(const _Float16* p0, const _Float16* p1) {
  v8h a = *(const v8h*)p0;
  v8h b = *(const v8h*)p1;
  v16h r;
#pragma unroll
  for (int i = 0; i < 8; ++i) { r[i] = a[i]; r[i + 8] = b[i]; }
  return r;
}

// ---------------------------------------------------------------------------
// Conversion / layout kernels
// ---------------------------------------------------------------------------

// x f32 [B,N,D] -> xh f16 [B*N, D] and xhT f16 [B, D, N]
__global__ void k_cvt_x(const float* __restrict__ x, _Float16* __restrict__ xh,
                        _Float16* __restrict__ xhT) {
  size_t i = (size_t)blockIdx.x * 256 + threadIdx.x;
  if (i >= (size_t)BATCH * NSEQ * DMODEL) return;
  float v = x[i];
  xh[i] = (_Float16)v;
  size_t d  = i % DMODEL;
  size_t bn = i / DMODEL;
  size_t n  = bn % NSEQ;
  size_t b  = bn / NSEQ;
  xhT[(b * DMODEL + d) * NSEQ + n] = (_Float16)v;
}

// transpose f32 [R,C] -> f16 [C,R]
__global__ void k_tr(const float* __restrict__ w, _Float16* __restrict__ wt,
                     int R, int C) {
  size_t i = (size_t)blockIdx.x * 256 + threadIdx.x;
  if (i >= (size_t)R * C) return;
  int r = (int)(i / C), c = (int)(i % C);
  wt[(size_t)c * R + r] = (_Float16)w[i];
}

// qk_tmp [B*N, 2*H*DH] -> q [B,H,N,DH] and khT [B,H,DH,N]
__global__ void k_scatter_qk(const _Float16* __restrict__ qk,
                             _Float16* __restrict__ q,
                             _Float16* __restrict__ khT) {
  size_t i = (size_t)blockIdx.x * 256 + threadIdx.x;
  if (i >= (size_t)BATCH * NSEQ * 2 * NHEAD * HDIM) return;
  int o = (int)(i & (2 * NHEAD * HDIM - 1));   // 0..2047
  size_t bn = i >> 11;
  int n = (int)(bn & (NSEQ - 1));
  int b = (int)(bn >> 11);
  _Float16 val = qk[i];
  if (o < NHEAD * HDIM) {
    int h = o >> 6, dh = o & 63;
    q[(((size_t)(b * NHEAD + h)) * NSEQ + n) * HDIM + dh] = val;
  } else {
    int oo = o - NHEAD * HDIM;
    int h = oo >> 6, dh = oo & 63;
    khT[(((size_t)(b * NHEAD + h)) * HDIM + dh) * NSEQ + n] = val;
  }
}

// v_tmp [B*N, H*DH] -> v [B,H,N,DH]
__global__ void k_scatter_v(const _Float16* __restrict__ vt,
                            _Float16* __restrict__ v) {
  size_t i = (size_t)blockIdx.x * 256 + threadIdx.x;
  if (i >= (size_t)BATCH * NSEQ * NHEAD * HDIM) return;
  int o = (int)(i & (NHEAD * HDIM - 1));
  size_t bn = i >> 10;
  int n = (int)(bn & (NSEQ - 1));
  int b = (int)(bn >> 11);
  int h = o >> 6, dh = o & 63;
  v[(((size_t)(b * NHEAD + h)) * NSEQ + n) * HDIM + dh] = vt[i];
}

// ---------------------------------------------------------------------------
// Generic f16 WMMA GEMM: C[M,N] = A[M,K] @ B[K,N]   (A,B row-major f16)
// MODE 0: store f16   MODE 1: store f32 (overwrite)   MODE 2: f32 accumulate
// Block: 256 threads (8 waves), tile 64x64, BK=32. Requires M%64==N%64==K%32==0.
// ---------------------------------------------------------------------------
template <int MODE>
__global__ __launch_bounds__(256) void gemm_f16(const _Float16* __restrict__ A,
                                                const _Float16* __restrict__ B,
                                                void* __restrict__ C,
                                                int M, int Nn, int K,
                                                long sAb, long sBb, long sCb) {
  __shared__ _Float16 As[64 * 40];   // padded stride 40 halves (80B, 16B aligned)
  __shared__ _Float16 Bs[32 * 72];   // padded stride 72 halves (144B, 16B aligned)

  const int tid  = threadIdx.x;
  const int wv   = tid >> 5;
  const int lane = tid & 31;
  const int half = lane >> 4;
  const int l16  = lane & 15;
  const int tm0  = blockIdx.y * 64;
  const int tn0  = blockIdx.x * 64;
  A += (size_t)blockIdx.z * sAb;
  B += (size_t)blockIdx.z * sBb;

  // wave -> two adjacent col tiles in one row tile (4x4 = 16 tiles / 8 waves)
  const int r  = (2 * wv) >> 2;      // 0..3
  const int c0 = (2 * wv) & 3;       // c0 and c0+1 (c0 in {0,2})

  v8f acc0 = {}, acc1 = {};

  const int ia = tid * 8;
  const int ar = ia >> 5, ac = ia & 31;  // A tile 64x32
  const int br = ia >> 6, bc = ia & 63;  // B tile 32x64

  for (int k0 = 0; k0 < K; k0 += 32) {
    *(v8h*)(As + ar * 40 + ac) = *(const v8h*)(A + (size_t)(tm0 + ar) * K + k0 + ac);
    *(v8h*)(Bs + br * 72 + bc) = *(const v8h*)(B + (size_t)(k0 + br) * Nn + tn0 + bc);
    if (k0 + 32 < K) {  // pull next tiles toward GL2 (global_prefetch_b8)
      __builtin_prefetch(A + (size_t)(tm0 + ar) * K + k0 + 32 + ac, 0, 1);
      __builtin_prefetch(B + (size_t)(k0 + 32 + br) * Nn + tn0 + bc, 0, 1);
    }
    __syncthreads();

    const _Float16* ap = As + (r * 16 + l16) * 40 + half * 8;
    v16h af = ld_frag16(ap, ap + 16);
    const _Float16* bp0 = Bs + c0 * 16 + l16;
    v16h bf0;
    for (int s = 0; s < 8; ++s) { bf0[s] = bp0[(half * 8 + s) * 72]; bf0[s + 8] = bp0[(16 + half * 8 + s) * 72]; }
    acc0 = __builtin_amdgcn_wmma_f32_16x16x32_f16(false, af, false, bf0,
                                                  (short)0, acc0, false, false);
    const _Float16* bp1 = Bs + (c0 + 1) * 16 + l16;
    v16h bf1;
    for (int s = 0; s < 8; ++s) { bf1[s] = bp1[(half * 8 + s) * 72]; bf1[s + 8] = bp1[(16 + half * 8 + s) * 72]; }
    acc1 = __builtin_amdgcn_wmma_f32_16x16x32_f16(false, af, false, bf1,
                                                  (short)0, acc1, false, false);
    __syncthreads();
  }

  // C fragment: vgpr i -> m = i + 8*half, lane -> n = l16
#pragma unroll
  for (int i = 0; i < 8; ++i) {
    int m  = tm0 + r * 16 + i + 8 * half;
    int n0 = tn0 + c0 * 16 + l16;
    int n1 = n0 + 16;
    if (MODE == 0) {
      _Float16* Ch = (_Float16*)C + (size_t)blockIdx.z * sCb;
      Ch[(size_t)m * Nn + n0] = (_Float16)acc0[i];
      Ch[(size_t)m * Nn + n1] = (_Float16)acc1[i];
    } else if (MODE == 1) {
      float* Cf = (float*)C + (size_t)blockIdx.z * sCb;
      Cf[(size_t)m * Nn + n0] = acc0[i];
      Cf[(size_t)m * Nn + n1] = acc1[i];
    } else {
      float* Cf = (float*)C + (size_t)blockIdx.z * sCb;
      Cf[(size_t)m * Nn + n0] += acc0[i];
      Cf[(size_t)m * Nn + n1] += acc1[i];
    }
  }
}

// ---------------------------------------------------------------------------
// Issue a 1-row TDM copy of `bytes/2` f16 elements (contiguous) from global
// `gaddr` into LDS byte offset `lds_byte_off`. D# per CDNA5 ISA.
// Caller must be a single wave; follow with s_wait_tensorcnt + barrier.
// ---------------------------------------------------------------------------
#if USE_TDM
static __device__ __forceinline__ void tdm_load_row_f16(unsigned lds_byte_off,
                                                        const void* gptr,
                                                        unsigned nelem) {
  unsigned long long ga = (unsigned long long)gptr;
  u32x4 g0;
  g0[0] = 1u;                                   // count=1, user descriptor
  g0[1] = lds_byte_off;                         // lds_addr
  g0[2] = (unsigned)(ga & 0xFFFFFFFFu);         // global_addr[31:0]
  g0[3] = (unsigned)((ga >> 32) & 0x01FFFFFFu)  // global_addr[56:32]
        | (2u << 30);                           // type=2 ("image")
  i32x8 g1;
  g1[0] = (int)(1u << 16);                      // workgroup_mask=0, data_size=1 (2B)
  g1[1] = (int)((nelem & 0xFFFFu) << 16);       // tensor_dim0[15:0] in [31:16]
  g1[2] = (int)(((nelem >> 16) & 0xFFFFu)       // tensor_dim0[31:16]
        | (1u << 16));                          // tensor_dim1 = 1
  g1[3] = (int)((nelem & 0xFFFFu) << 16);       // tile_dim0 = nelem
  g1[4] = 1;                                    // tile_dim1 = 1, tile_dim2 = 0
  g1[5] = (int)nelem;                           // tensor_dim0_stride[31:0]
  g1[6] = 0;                                    // stride0[47:32], stride1[15:0]
  g1[7] = 0;                                    // stride1[47:16]
  i32x4 gz = {};                                // groups 2/3 unused (<=2D tensor)
#if defined(__clang_major__) && (__clang_major__ >= 23)
  i32x8 gz8 = {};
  __builtin_amdgcn_tensor_load_to_lds(g0, g1, gz, gz, gz8, 0);
#else
  __builtin_amdgcn_tensor_load_to_lds(g0, g1, gz, gz, 0);
#endif
}
#endif

// ---------------------------------------------------------------------------
// Fused attention: one workgroup = (batch b, 32-row tile). Loops over all 16
// heads so the head-summed attention (asum) accumulates in LDS w/o atomics.
// Dynamic LDS (~261 KB) leans on gfx1250's 320 KB/WGP:
//   probs [32][2048] f16  (scores -> exp -> normalized probs, in place)
//   asml  [32][2048] f16  (sum_h w_h * attn_h)
//   qt    [32][64]   f16  (staged via TENSOR_LOAD_TO_LDS)
//   part[256], rowmax[32], rowsum[32] f32
// ---------------------------------------------------------------------------
__global__ __launch_bounds__(256) void attn_kernel(
    const _Float16* __restrict__ q,    // [B,H,N,DH]
    const _Float16* __restrict__ khT,  // [B,H,DH,N]
    const _Float16* __restrict__ v,    // [B,H,N,DH]
    const _Float16* __restrict__ xxt,  // [B,N,N] f16
    const float* __restrict__ wkwq,    // [H]
    const float* __restrict__ wvwo,    // [H]
    _Float16* __restrict__ y,          // [B,N,H*DH]
    _Float16* __restrict__ asum)       // [B,N,N]
{
  extern __shared__ __align__(16) char smem_raw[];
  _Float16* probs = (_Float16*)smem_raw;          // 32*2048
  _Float16* asml  = probs + MT * NSEQ;            // 32*2048
  _Float16* qt    = asml + MT * NSEQ;             // 32*64
  float* part     = (float*)(qt + MT * HDIM);     // 256
  float* rowmax   = part + 256;                   // 32
  float* rowsum   = rowmax + 32;                  // 32

  const int b    = blockIdx.y;
  const int m0   = blockIdx.x * MT;
  const int tid  = threadIdx.x;
  const int wv   = tid >> 5;
  const int lane = tid & 31;
  const int half = lane >> 4;
  const int l16  = lane & 15;

  // zero the head-summed attention accumulator
  for (int i = tid * 8; i < MT * NSEQ; i += 256 * 8) {
    v8h z = {};
    *(v8h*)(asml + i) = z;
  }
  __syncthreads();

  const _Float16* xxb = xxt + (size_t)b * NSEQ * NSEQ;

#if USE_TDM
  // dynamic-LDS byte offset of qt (dynamic LDS starts after static allocs)
  const unsigned qt_lds =
      __builtin_amdgcn_groupstaticsize() +
      (unsigned)((const char*)qt - (const char*)smem_raw);
#endif

  for (int h = 0; h < NHEAD; ++h) {
    const float ch = wkwq[h];
    const float wh = wvwo[h];

    // stage q tile 32x64 (4 KB contiguous) into LDS
    const _Float16* qs = q + (((size_t)(b * NHEAD + h)) * NSEQ + m0) * HDIM;
#if USE_TDM
    if (wv == 0) {
      tdm_load_row_f16(qt_lds, qs, MT * HDIM);
      __builtin_amdgcn_s_wait_tensorcnt(0);
    }
#else
    {
      int i = tid * 8;
      *(v8h*)(qt + i) = *(const v8h*)(qs + i);
    }
#endif
    __syncthreads();

    // ---- Phase A: S = q @ k^T * SCALE  (wave w -> 256-col strip)
    const _Float16* kb = khT + ((size_t)(b * NHEAD + h)) * HDIM * NSEQ;
    for (int ct = 0; ct < 16; ++ct) {
      const int col0 = wv * 256 + ct * 16;
      v8f s0 = {}, s1 = {};
#pragma unroll
      for (int ks = 0; ks < 2; ++ks) {
        const _Float16* bp = kb + (size_t)(ks * 32) * NSEQ + col0 + l16;
        v16h bf;
        for (int s = 0; s < 8; ++s) { bf[s] = bp[(size_t)(half * 8 + s) * NSEQ]; bf[s + 8] = bp[(size_t)(16 + half * 8 + s) * NSEQ]; }
        const _Float16* a0 = qt + l16 * HDIM + ks * 32 + half * 8;
        v16h af0 = ld_frag16(a0, a0 + 16);
        s0 = __builtin_amdgcn_wmma_f32_16x16x32_f16(false, af0, false, bf,
                                                    (short)0, s0, false, false);
        const _Float16* a1 = qt + (16 + l16) * HDIM + ks * 32 + half * 8;
        v16h af1 = ld_frag16(a1, a1 + 16);
        s1 = __builtin_amdgcn_wmma_f32_16x16x32_f16(false, af1, false, bf,
                                                    (short)0, s1, false, false);
      }
#pragma unroll
      for (int i = 0; i < 8; ++i) {
        int ml = i + 8 * half;
        int n  = col0 + l16;
        probs[ml * NSEQ + n]        = (_Float16)(s0[i] * SCALE);
        probs[(16 + ml) * NSEQ + n] = (_Float16)(s1[i] * SCALE);
      }
    }
    __syncthreads();

    // ---- bias + softmax over each row (8 threads/row, 256 cols/thread).
    // xxt bias added here: contiguous v8h loads instead of per-element gathers.
    {
      int r = tid >> 3, seg = tid & 7;
      _Float16* pr = probs + r * NSEQ + seg * 256;
      const _Float16* xr = xxb + (size_t)(m0 + r) * NSEQ + seg * 256;
      float mx = -3.0e38f;
      for (int c = 0; c < 256; c += 8) {
        v8h sv = *(const v8h*)(pr + c);
        v8h xv = *(const v8h*)(xr + c);
#pragma unroll
        for (int j = 0; j < 8; ++j) {
          float s = (float)sv[j] + ch * (float)xv[j];
          sv[j] = (_Float16)s;
          mx = fmaxf(mx, s);
        }
        *(v8h*)(pr + c) = sv;   // biased score back in place
      }
      part[tid] = mx;
      __syncthreads();
      if (tid < 32) {
        float m = -3.0e38f;
        for (int s = 0; s < 8; ++s) m = fmaxf(m, part[tid * 8 + s]);
        rowmax[tid] = m;
      }
      __syncthreads();

      float rm = rowmax[r];
      float sum = 0.f;
      for (int c = 0; c < 256; c += 8) {
        v8h sv = *(const v8h*)(pr + c);
#pragma unroll
        for (int j = 0; j < 8; ++j) {
          float p = __expf((float)sv[j] - rm);
          sv[j] = (_Float16)p;
          sum += p;
        }
        *(v8h*)(pr + c) = sv;
      }
      part[tid] = sum;
      __syncthreads();
      if (tid < 32) {
        float s2 = 0.f;
        for (int s = 0; s < 8; ++s) s2 += part[tid * 8 + s];
        rowsum[tid] = s2;
      }
      __syncthreads();

      float inv = 1.0f / rowsum[r];
      _Float16* aw = asml + r * NSEQ + seg * 256;
      for (int c = 0; c < 256; c += 8) {
        v8h sv = *(const v8h*)(pr + c);
        v8h av = *(const v8h*)(aw + c);
#pragma unroll
        for (int j = 0; j < 8; ++j) {
          float pn = (float)sv[j] * inv;
          sv[j] = (_Float16)pn;                       // normalized prob
          av[j] = (_Float16)((float)av[j] + wh * pn); // head-weighted sum
        }
        *(v8h*)(pr + c) = sv;
        *(v8h*)(aw + c) = av;
      }
    }
    __syncthreads();

    // ---- Phase B: O = P @ V, 8 output tiles (2x4 of 16x16), one per wave
    {
      const int mt = wv >> 2;   // 0..1
      const int nt = wv & 3;    // 0..3
      const _Float16* vb = v + ((size_t)(b * NHEAD + h)) * NSEQ * HDIM + nt * 16;
      v8f acc = {};
      for (int ks = 0; ks < NSEQ / 32; ++ks) {
        const _Float16* ap = probs + (mt * 16 + l16) * NSEQ + ks * 32 + half * 8;
        v16h af = ld_frag16(ap, ap + 16);
        const _Float16* bp = vb + (size_t)(ks * 32) * HDIM + l16;
        v16h bf;
        for (int s = 0; s < 8; ++s) { bf[s] = bp[(size_t)(half * 8 + s) * HDIM]; bf[s + 8] = bp[(size_t)(16 + half * 8 + s) * HDIM]; }
        acc = __builtin_amdgcn_wmma_f32_16x16x32_f16(false, af, false, bf,
                                                     (short)0, acc, false, false);
      }
      _Float16* yb = y + ((size_t)b * NSEQ + m0 + mt * 16) * (NHEAD * HDIM) +
                     h * HDIM + nt * 16 + l16;
#pragma unroll
      for (int i = 0; i < 8; ++i) {
        int ml = i + 8 * half;
        yb[(size_t)ml * (NHEAD * HDIM)] = (_Float16)acc[i];
      }
    }
    __syncthreads();
  }

  // flush head-summed attention tile to global
  {
    _Float16* ag = asum + ((size_t)b * NSEQ + m0) * NSEQ;
    for (int i = tid * 8; i < MT * NSEQ; i += 256 * 8) {
      int r = i / NSEQ, c = i % NSEQ;
      *(v8h*)(ag + (size_t)r * NSEQ + c) = *(const v8h*)(asml + i);
    }
  }
}

// ---------------------------------------------------------------------------
// Host launcher
// ---------------------------------------------------------------------------
extern "C" void kernel_launch(void* const* d_in, const int* in_sizes, int n_in,
                              void* d_out, int out_size, void* d_ws, size_t ws_size,
                              hipStream_t stream) {
  (void)in_sizes; (void)n_in; (void)out_size; (void)ws_size;
  const float* x     = (const float*)d_in[0];  // [B,N,D]
  const float* w_qk  = (const float*)d_in[1];  // [2*H*DH, D]
  const float* w_v   = (const float*)d_in[2];  // [H*DH, D]
  const float* w_out = (const float*)d_in[3];  // [D, H*DH]
  const float* wkwq  = (const float*)d_in[4];  // [H]
  const float* wvwo  = (const float*)d_in[5];  // [H]
  float* out = (float*)d_out;                  // [B,N,D] f32

  // workspace carve-up (all regions 256B aligned). Total ~88 MB.
  char* ws = (char*)d_ws;
  size_t off = 0;
  auto alloc = [&](size_t bytes) {
    size_t o = off;
    off += (bytes + 255) & ~(size_t)255;
    return o;
  };
  _Float16* xh     = (_Float16*)(ws + alloc((size_t)BATCH * NSEQ * DMODEL * 2));          // 8 MB
  _Float16* xhT    = (_Float16*)(ws + alloc((size_t)BATCH * DMODEL * NSEQ * 2));          // 8 MB
  _Float16* wqkT   = (_Float16*)(ws + alloc((size_t)DMODEL * 2 * NHEAD * HDIM * 2));      // 4 MB
  _Float16* wvT    = (_Float16*)(ws + alloc((size_t)DMODEL * NHEAD * HDIM * 2));          // 2 MB
  _Float16* woT    = (_Float16*)(ws + alloc((size_t)NHEAD * HDIM * DMODEL * 2));          // 2 MB
  _Float16* qk_tmp = (_Float16*)(ws + alloc((size_t)BATCH * NSEQ * NSEQ * 2));            // 16 MB (reused as asum)
  _Float16* v_tmp  = (_Float16*)(ws + alloc((size_t)BATCH * NSEQ * NHEAD * HDIM * 2));    // 8 MB (reused as y)
  _Float16* qbuf   = (_Float16*)(ws + alloc((size_t)BATCH * NHEAD * NSEQ * HDIM * 2));    // 8 MB
  _Float16* khT    = (_Float16*)(ws + alloc((size_t)BATCH * NHEAD * HDIM * NSEQ * 2));    // 8 MB
  _Float16* vbuf   = (_Float16*)(ws + alloc((size_t)BATCH * NHEAD * NSEQ * HDIM * 2));    // 8 MB
  _Float16* xxt    = (_Float16*)(ws + alloc((size_t)BATCH * NSEQ * NSEQ * 2));            // 16 MB
  _Float16* asum   = qk_tmp;  // qk_tmp dead after scatter
  _Float16* ybuf   = v_tmp;   // v_tmp dead after scatter

  // 1) convert / transpose
  k_cvt_x<<<(BATCH * NSEQ * DMODEL + 255) / 256, 256, 0, stream>>>(x, xh, xhT);
  k_tr<<<(2 * NHEAD * HDIM * DMODEL + 255) / 256, 256, 0, stream>>>(w_qk, wqkT, 2 * NHEAD * HDIM, DMODEL);
  k_tr<<<(NHEAD * HDIM * DMODEL + 255) / 256, 256, 0, stream>>>(w_v, wvT, NHEAD * HDIM, DMODEL);
  k_tr<<<(DMODEL * NHEAD * HDIM + 255) / 256, 256, 0, stream>>>(w_out, woT, DMODEL, NHEAD * HDIM);

  // 2) projections: qk = xh @ wqkT  [4096 x 2048 x 1024],  v = xh @ wvT
  gemm_f16<0><<<dim3(2 * NHEAD * HDIM / 64, BATCH * NSEQ / 64, 1), 256, 0, stream>>>(
      xh, wqkT, qk_tmp, BATCH * NSEQ, 2 * NHEAD * HDIM, DMODEL, 0, 0, 0);
  gemm_f16<0><<<dim3(NHEAD * HDIM / 64, BATCH * NSEQ / 64, 1), 256, 0, stream>>>(
      xh, wvT, v_tmp, BATCH * NSEQ, NHEAD * HDIM, DMODEL, 0, 0, 0);

  // 3) xxt[b] = xh[b] @ xhT[b]  [2048 x 2048 x 1024], batched over b
  gemm_f16<0><<<dim3(NSEQ / 64, NSEQ / 64, BATCH), 256, 0, stream>>>(
      xh, xhT, xxt, NSEQ, NSEQ, DMODEL,
      (long)NSEQ * DMODEL, (long)DMODEL * NSEQ, (long)NSEQ * NSEQ);

  // 4) scatter into per-head layouts (q [B,H,N,DH], khT [B,H,DH,N], v [B,H,N,DH])
  k_scatter_qk<<<(BATCH * NSEQ * 2 * NHEAD * HDIM + 255) / 256, 256, 0, stream>>>(qk_tmp, qbuf, khT);
  k_scatter_v<<<(BATCH * NSEQ * NHEAD * HDIM + 255) / 256, 256, 0, stream>>>(v_tmp, vbuf);

  // 5) fused attention (261 KB dynamic LDS; gfx1250 WGP has 320 KB)
  size_t shmem = (size_t)MT * NSEQ * 2 * 2   // probs + asml
               + (size_t)MT * HDIM * 2       // qt
               + (256 + 32 + 32) * 4;        // reductions
  attn_kernel<<<dim3(NSEQ / MT, BATCH), 256, shmem, stream>>>(
      qbuf, khT, vbuf, xxt, wkwq, wvwo, ybuf, asum);

  // 6) out = y @ woT  (f32 overwrite of poisoned d_out)
  gemm_f16<1><<<dim3(DMODEL / 64, BATCH * NSEQ / 64, 1), 256, 0, stream>>>(
      ybuf, woT, out, BATCH * NSEQ, DMODEL, NHEAD * HDIM, 0, 0, 0);

  // 7) out += asum[b] @ xh[b]   [2048 x 1024 x 2048], batched over b
  gemm_f16<2><<<dim3(DMODEL / 64, NSEQ / 64, BATCH), 256, 0, stream>>>(
      asum, xh, out, NSEQ, DMODEL, NSEQ,
      (long)NSEQ * NSEQ, (long)NSEQ * DMODEL, (long)NSEQ * DMODEL);
}
